// KernelAttention_14130442404180
// MI455X (gfx1250) — matrix operand order repaired and checked by reading, not compiled, not run on hardware
//
#include <hip/hip_runtime.h>
#include <hip/hip_bf16.h>
#include <math.h>

// ---------------------------------------------------------------------------
// RBF-kernel attention for MI455X (gfx1250, wave32, WMMA).
//  B=8, Q=K=4096, D=256, d=64.  Outputs: out [B,Q,64] then attn [B,Q,K] (f32).
//
//  Roofline: attn writeout (512 MB, write-once) dominates -> non-temporal
//  stores; all big GEMMs via v_wmma_f32_16x16x32_bf16; per-q-tile scores kept
//  entirely in 256 KB LDS so normalized attn is written to HBM exactly once.
// ---------------------------------------------------------------------------

typedef __attribute__((ext_vector_type(16))) __bf16 v16bf;
typedef __attribute__((ext_vector_type(8)))  float  v8f;
typedef __attribute__((ext_vector_type(4)))  float  v4f;

union BF16Frag {
    uint4  u4[2];   // 2 x 16B = 32B = 16 bf16
    v16bf  v;
    __bf16 e[16];
};

static constexpr int B_    = 8;
static constexpr int Q_    = 4096;
static constexpr int K_    = 4096;
static constexpr int D_    = 256;
static constexpr int d_    = 64;
static constexpr int NROWS = B_ * Q_;   // 32768

// ---------------------------------------------------------------------------
// Kernel 1: projections.  x[32768,256] @ W[256,64] + b  for q / k / v.
//  q,k -> bf16 row-major + row squared-norms.   v -> bf16 TRANSPOSED per batch
//  ([B][64][4096]) so attn@v B-fragments are contiguous along K.
// ---------------------------------------------------------------------------
__global__ void __launch_bounds__(256)
proj_kernel(const float* __restrict__ qs, const float* __restrict__ ks,
            const float* __restrict__ vs,
            const float* __restrict__ Wq, const float* __restrict__ bq,
            const float* __restrict__ Wk, const float* __restrict__ bk,
            const float* __restrict__ Wv, const float* __restrict__ bv,
            __bf16* __restrict__ q_bf, __bf16* __restrict__ k_bf,
            __bf16* __restrict__ vT_bf,
            float* __restrict__ q2, float* __restrict__ k2)
{
    __shared__ float Wlds[128 * 64];      // 32 KB: half of W at a time
    __shared__ float rowpart[64][4];

    const int m = blockIdx.y;             // 0:q 1:k 2:v
    const float* X; const float* W; const float* bias;
    if (m == 0)      { X = qs; W = Wq; bias = bq; }
    else if (m == 1) { X = ks; W = Wk; bias = bk; }
    else             { X = vs; W = Wv; bias = bv; }

    const int  t     = threadIdx.x;
    const int  row_l = t >> 2;            // 0..63
    const int  cseg  = t & 3;             // 16-col segment
    const long rowg  = (long)blockIdx.x * 64 + row_l;

    float acc[16];
    #pragma unroll
    for (int i = 0; i < 16; ++i) acc[i] = bias[cseg * 16 + i];

    for (int chunk = 0; chunk < 2; ++chunk) {
        __syncthreads();
        for (int i = t; i < 128 * 64; i += 256)
            Wlds[i] = W[chunk * 128 * 64 + i];
        __syncthreads();
        const float* xrow = X + rowg * D_ + chunk * 128;
        for (int kk = 0; kk < 128; ++kk) {
            const float xv = xrow[kk];
            #pragma unroll
            for (int i = 0; i < 16; ++i)
                acc[i] += xv * Wlds[kk * 64 + cseg * 16 + i];
        }
    }

    if (m < 2) {
        __bf16* outp = (m == 0 ? q_bf : k_bf) + rowg * d_ + cseg * 16;
        float ss = 0.f;
        #pragma unroll
        for (int i = 0; i < 16; ++i) {
            outp[i] = (__bf16)acc[i];
            ss += acc[i] * acc[i];
        }
        rowpart[row_l][cseg] = ss;
        __syncthreads();
        if (t < 64) {
            const float n = rowpart[t][0] + rowpart[t][1] +
                            rowpart[t][2] + rowpart[t][3];
            const long rg = (long)blockIdx.x * 64 + t;
            (m == 0 ? q2 : k2)[rg] = n;
        }
    } else {
        const int b  = (int)(rowg >> 12);       // / 4096
        const int rk = (int)(rowg & 4095);
        #pragma unroll
        for (int i = 0; i < 16; ++i) {
            const int col = cseg * 16 + i;
            vT_bf[((long)b * 64 + col) * K_ + rk] = (__bf16)acc[i];
        }
    }
}

// ---------------------------------------------------------------------------
// Kernel 2: per (batch, 16-row q tile), 256 threads = 8 waves (2 per SIMD32;
// the 268 KB LDS already limits to 1 workgroup/WGP, so 8 waves maximizes
// latency hiding for the global fragment loads between WMMAs):
//   phase A: S = q·k^T via v_wmma_f32_16x16x32_bf16, e = exp(-max(d2,0)/8),
//            masked by valid_len, staged unnormalized in a 256KB LDS buffer.
//   phase B: LDS row-sum reduction -> 1/rowsum.
//   phase C: coalesced NON-TEMPORAL float4 write of normalized attn
//            (write-once 512MB stream; keep L2 for the reused k/v tiles).
//   phase D: ctx = (e/rowsum) @ v via WMMA; 8 waves = 4 N-tiles x 2 K-halves,
//            cross-wave partial reduction through LDS.
// ---------------------------------------------------------------------------
__global__ void __launch_bounds__(256)
attn_kernel(const __bf16* __restrict__ q_bf, const __bf16* __restrict__ k_bf,
            const __bf16* __restrict__ vT_bf,
            const float* __restrict__ q2, const float* __restrict__ k2,
            const int* __restrict__ valid_lens,
            float* __restrict__ attn_out, float* __restrict__ ctx)
{
    extern __shared__ char smem_raw[];
    float* ebuf    = (float*)smem_raw;    // 16 * 4096 f32 = 256 KB
    float* q2s     = ebuf + 16 * 4096;    // 16
    float* invs    = q2s + 16;            // 16
    float* rowpart = invs + 16;           // 16 * 16 = 256
    float* cred    = rowpart + 256;       // 4 * 16 * 16 = 1024 (phase D partials)

    const int t    = threadIdx.x;
    const int wv   = t >> 5;              // wave 0..7
    const int lane = t & 31;
    const int nIdx = lane & 15;           // N / row-in-half per WMMA layouts
    const int hf   = lane >> 4;           // lane half (0/1)

    const int  blk     = blockIdx.x;      // 0..2047
    const int  b       = blk >> 8;        // 256 q-tiles per batch
    const int  qt      = blk & 255;
    const long rowbase = (long)b * Q_ + qt * 16;
    const int  vlen    = valid_lens[b];

    if (t < 16) q2s[t] = q2[rowbase + t];
    __syncthreads();

    // ---- A fragments for q (16x32 bf16), two for d=64 ----
    // Lane holds row = lane&15; K runs: f*32 + hf*16 + {0..15} contig.
    const __bf16* qrow = q_bf + (rowbase + nIdx) * d_;
    BF16Frag aq[2];
    #pragma unroll
    for (int f = 0; f < 2; ++f) {
        aq[f].u4[0] = *(const uint4*)(qrow + f * 32 + hf * 16);
        aq[f].u4[1] = *(const uint4*)(qrow + f * 32 + hf * 16 + 8);
    }
    float q2r[8];
    #pragma unroll
    for (int r = 0; r < 8; ++r) q2r[r] = q2s[r + (hf << 3)];

    const float*  k2b     = k2 + (long)b * K_;
    const __bf16* kb_base = k_bf + (long)b * K_ * (long)d_;

    // ---- phase A: scores over all 4096 keys, 16-key tiles, 8 waves strided -
    for (int tile = wv; tile < K_ / 16; tile += 8) {
        const int kb = tile * 16;
        // B fragment (32x16): lane col = lane&15, K = hf*16 + {0..15} contig.
        const __bf16* krow = kb_base + (long)(kb + nIdx) * d_;
        BF16Frag bk0, bk1;
        bk0.u4[0] = *(const uint4*)(krow + 0 * 32 + hf * 16);
        bk0.u4[1] = *(const uint4*)(krow + 0 * 32 + hf * 16 + 8);
        bk1.u4[0] = *(const uint4*)(krow + 1 * 32 + hf * 16);
        bk1.u4[1] = *(const uint4*)(krow + 1 * 32 + hf * 16 + 8);

        v8f c = {};
        c = __builtin_amdgcn_wmma_f32_16x16x32_bf16(false, aq[0].v, false, bk0.v,
                                                    (short)0, c, false, false);
        c = __builtin_amdgcn_wmma_f32_16x16x32_bf16(false, aq[1].v, false, bk1.v,
                                                    (short)0, c, false, false);

        const int   col   = kb + nIdx;
        const float k2v   = k2b[col];
        const bool  valid = col < vlen;
        #pragma unroll
        for (int r = 0; r < 8; ++r) {
            const int M = r + (hf << 3);          // C layout: M = r + 8*half
            float d2 = q2r[r] + k2v - 2.0f * c[r];
            d2 = fmaxf(d2, 0.0f);
            const float e = valid ? __expf(-d2 * 0.125f) : 0.0f;
            ebuf[M * 4096 + col] = e;
        }
    }
    __syncthreads();

    // ---- phase B: row sums -> 1/sum (256 threads: 16 rows x 16 segments) ---
    {
        const int row = t >> 4, seg = t & 15;
        const float* p = ebuf + row * 4096 + seg * 256;
        float s = 0.f;
        for (int i = 0; i < 256; ++i) s += p[i];
        rowpart[row * 16 + seg] = s;
    }
    __syncthreads();
    if (t < 16) {
        float s = 0.f;
        #pragma unroll
        for (int i = 0; i < 16; ++i) s += rowpart[t * 16 + i];
        invs[t] = 1.0f / s;
    }
    __syncthreads();

    // ---- phase C: non-temporal write of normalized attn ----
    // LDS flat layout [16*4096] == global layout of this block's attn slice.
    {
        float* attn_blk = attn_out + rowbase * (long)K_;  // contiguous 16*4096
        for (int i = t * 4; i < 16 * 4096; i += 256 * 4) {
            v4f v = *(const v4f*)(ebuf + i);
            v *= invs[i >> 12];                           // row = i / 4096
            __builtin_nontemporal_store(v, (v4f*)(attn_blk + i));
        }
    }

    // ---- phase D: ctx = (e * inv) @ v ----
    // wave -> N tile (wv&3), K half (wv>>2): keys [kh*2048, kh*2048+2048).
    const int     ntile = wv & 3;
    const int     kh    = wv >> 2;
    const int     ncol  = ntile * 16 + nIdx;
    const __bf16* vTrow = vT_bf + ((long)b * 64 + ncol) * K_;
    const float   ascal = invs[nIdx];                     // A row = lane&15
    v8f cacc = {};
    for (int kt = kh * 64; kt < kh * 64 + 64; ++kt) {
        const int k0 = kt * 32;
        BF16Frag ae;
        const float* erow = ebuf + nIdx * 4096 + k0;
        #pragma unroll
        for (int vvp = 0; vvp < 8; ++vvp) {
            const int base = hf * 16 + vvp * 2;
            ae.e[vvp * 2]     = (__bf16)(erow[base]     * ascal);
            ae.e[vvp * 2 + 1] = (__bf16)(erow[base + 1] * ascal);
        }
        BF16Frag bvf;
        bvf.u4[0] = *(const uint4*)(vTrow + k0 + hf * 16);
        bvf.u4[1] = *(const uint4*)(vTrow + k0 + hf * 16 + 8);
        cacc = __builtin_amdgcn_wmma_f32_16x16x32_bf16(false, ae.v, false, bvf.v,
                                                       (short)0, cacc, false, false);
    }
    // cross-wave reduction: waves 4..7 stage partials, waves 0..3 combine.
    if (kh == 1) {
        #pragma unroll
        for (int r = 0; r < 8; ++r) {
            const int M = r + (hf << 3);
            cred[ntile * 256 + M * 16 + nIdx] = cacc[r];
        }
    }
    __syncthreads();
    if (kh == 0) {
        #pragma unroll
        for (int r = 0; r < 8; ++r) {
            const int M = r + (hf << 3);
            const float val = cacc[r] + cred[ntile * 256 + M * 16 + nIdx];
            ctx[(rowbase + M) * 64 + ncol] = val;
        }
    }
}

// ---------------------------------------------------------------------------
// Kernel 3: out = ctx @ Wo + bo   ([32768,64] @ [64,64], trivial)
// ---------------------------------------------------------------------------
__global__ void __launch_bounds__(256)
out_kernel(const float* __restrict__ ctx, const float* __restrict__ Wo,
           const float* __restrict__ bo, float* __restrict__ out)
{
    __shared__ float Wl[64 * 64];
    const int t = threadIdx.x;
    for (int i = t; i < 64 * 64; i += 256) Wl[i] = Wo[i];
    __syncthreads();

    const int  row_l = t >> 2, cseg = t & 3;
    const long rowg  = (long)blockIdx.x * 64 + row_l;
    const float* crow = ctx + rowg * 64;

    float acc[16];
    #pragma unroll
    for (int i = 0; i < 16; ++i) acc[i] = bo[cseg * 16 + i];
    for (int kk = 0; kk < 64; ++kk) {
        const float cv = crow[kk];
        #pragma unroll
        for (int i = 0; i < 16; ++i)
            acc[i] += cv * Wl[kk * 64 + cseg * 16 + i];
    }
    #pragma unroll
    for (int i = 0; i < 16; ++i)
        out[rowg * 64 + cseg * 16 + i] = acc[i];
}

// ---------------------------------------------------------------------------
extern "C" void kernel_launch(void* const* d_in, const int* in_sizes, int n_in,
                              void* d_out, int out_size, void* d_ws, size_t ws_size,
                              hipStream_t stream)
{
    (void)in_sizes; (void)n_in; (void)out_size; (void)ws_size;

    const float* qs         = (const float*)d_in[0];
    const float* ks         = (const float*)d_in[1];
    const float* vs         = (const float*)d_in[2];
    const int*   valid_lens = (const int*)  d_in[3];
    const float* Wq         = (const float*)d_in[4];
    const float* bq         = (const float*)d_in[5];
    const float* Wk         = (const float*)d_in[6];
    const float* bk         = (const float*)d_in[7];
    const float* Wv         = (const float*)d_in[8];
    const float* bv         = (const float*)d_in[9];
    const float* Wo         = (const float*)d_in[10];
    const float* bo         = (const float*)d_in[11];

    char*  ws  = (char*)d_ws;
    size_t off = 0;
    __bf16* q_bf  = (__bf16*)(ws + off); off += (size_t)NROWS * 64 * 2;   // 4 MB
    __bf16* k_bf  = (__bf16*)(ws + off); off += (size_t)NROWS * 64 * 2;   // 4 MB
    __bf16* vT_bf = (__bf16*)(ws + off); off += (size_t)NROWS * 64 * 2;   // 4 MB
    float*  q2    = (float*) (ws + off); off += (size_t)NROWS * 4;        // 128 KB
    float*  k2    = (float*) (ws + off); off += (size_t)NROWS * 4;        // 128 KB
    float*  ctx   = (float*) (ws + off); off += (size_t)NROWS * 64 * 4;   // 8 MB

    float* out  = (float*)d_out;
    float* attn = out + (size_t)NROWS * 64;   // tuple: (out, attn) concatenated

    proj_kernel<<<dim3(NROWS / 64, 3), 256, 0, stream>>>(
        qs, ks, vs, Wq, bq, Wk, bk, Wv, bv, q_bf, k_bf, vT_bf, q2, k2);

    const size_t smem2 = (size_t)(16 * 4096 + 16 + 16 + 256 + 1024) * sizeof(float);
    attn_kernel<<<dim3(B_ * (Q_ / 16)), 256, smem2, stream>>>(
        q_bf, k_bf, vT_bf, q2, k2, valid_lens, attn, ctx);

    out_kernel<<<dim3(NROWS / 64), 256, 0, stream>>>(ctx, Wo, bo, out);
}
